// AttentionHead_15058155340112
// MI455X (gfx1250) — compile-verified, matrix operand order checked by reading, not run on hardware
//
#include <hip/hip_runtime.h>
#include <hip/hip_bf16.h>

// ---------------------------------------------------------------------------
// AttentionHead for MI455X (gfx1250, wave32, WMMA bf16 16x16x32, f32 accum)
//
//   Pass 0: transpose+convert Wq/Wk/Wv (f32 [1024,128]) -> Wt (bf16 [3][128][1024])
//   Pass 1: QKV GEMM  x[16384,1024]f32 x Wt -> Q,K bf16 [BT,128]; V stored
//           transposed as Vt bf16 [B][128][2048]  (B-fragment friendly)
//   Pass 2: per-wave flash attention with causal mask + online softmax.
// ---------------------------------------------------------------------------

typedef __attribute__((ext_vector_type(16))) __bf16 bf16x16;
typedef __attribute__((ext_vector_type(8)))  __bf16 bf16x8;
typedef __attribute__((ext_vector_type(8)))  float  f32x8;

#define B_   8
#define T_   2048
#define DIN  1024
#define DOUT 128
#define BT   (B_ * T_)

#define WMMA_BF16(a, b, c) \
    __builtin_amdgcn_wmma_f32_16x16x32_bf16(false, (a), false, (b), (short)0, (c), false, false)

// ---------------------------------------------------------------------------
// Pass 0: Wt[mat][n][k] = (bf16) W[mat][k][n]
// ---------------------------------------------------------------------------
__global__ void prep_weights_kernel(const float* __restrict__ Wq,
                                    const float* __restrict__ Wk,
                                    const float* __restrict__ Wv,
                                    __bf16* __restrict__ Wt) {
    int idx = blockIdx.x * blockDim.x + threadIdx.x;
    if (idx >= 3 * DIN * DOUT) return;
    int mat = idx / (DIN * DOUT);
    int rem = idx - mat * (DIN * DOUT);
    int k = rem / DOUT;
    int n = rem - k * DOUT;
    const float* W = (mat == 0) ? Wq : (mat == 1) ? Wk : Wv;
    Wt[(size_t)mat * DIN * DOUT + (size_t)n * DIN + k] = (__bf16)W[rem];
}

// ---------------------------------------------------------------------------
// Pass 1: QKV projection GEMM.
//   block = 256 threads = 8 waves; wave handles a 16-row x 128-col tile.
//   grid  = (BT/128, 3)   blockIdx.y selects Wq/Wk/Wv.
// ---------------------------------------------------------------------------
__global__ __launch_bounds__(256)
void qkv_gemm_kernel(const float* __restrict__ x,
                     const __bf16* __restrict__ Wt_all,
                     __bf16* __restrict__ Qb,
                     __bf16* __restrict__ Kb,
                     __bf16* __restrict__ Vt) {
    const int lane = threadIdx.x & 31;
    const int wave = threadIdx.x >> 5;
    const int mat  = blockIdx.y;
    const __bf16* __restrict__ Wt = Wt_all + (size_t)mat * DIN * DOUT; // [DOUT][DIN]

    const int row0 = (blockIdx.x * 8 + wave) * 16;   // 16-row output tile
    const int half = lane >> 4;                      // 0 or 1
    const int lrow = lane & 15;
    const int koff = half ? 8 : 0;                   // A/B fragment K striping

    f32x8 acc[8];
#pragma unroll
    for (int nt = 0; nt < 8; ++nt)
#pragma unroll
        for (int r = 0; r < 8; ++r) acc[nt][r] = 0.0f;

    const float* __restrict__ xrow = x + (size_t)(row0 + lrow) * DIN;

    for (int kb = 0; kb < DIN; kb += 32) {
        // A fragment: x row, convert f32 -> bf16 (two contiguous runs of 8)
        bf16x16 a;
        const float* xp = xrow + kb + koff;
#pragma unroll
        for (int i = 0; i < 8; ++i) {
            a[i]     = (__bf16)xp[i];
            a[8 + i] = (__bf16)xp[16 + i];
        }
#pragma unroll
        for (int nt = 0; nt < 8; ++nt) {
            // B fragment: column (nt*16+lrow) of W == row of Wt, contiguous in k
            const __bf16* wp = Wt + (size_t)(nt * 16 + lrow) * DIN + kb + koff;
            bf16x16 bfrag;
#pragma unroll
            for (int i = 0; i < 8; ++i) {
                bfrag[i]     = wp[i];
                bfrag[8 + i] = wp[16 + i];
            }
            acc[nt] = WMMA_BF16(a, bfrag, acc[nt]);
        }
    }

    if (mat == 2) {
        // V: store transposed Vt[b][e][t]; each lane's 8 rows are contiguous t
        const int t0   = (row0 & (T_ - 1)) + half * 8;
        const int bidx = row0 / T_;
#pragma unroll
        for (int nt = 0; nt < 8; ++nt) {
            bf16x8 v;
#pragma unroll
            for (int r = 0; r < 8; ++r) v[r] = (__bf16)acc[nt][r];
            __bf16* dst = Vt + ((size_t)bidx * DOUT + nt * 16 + lrow) * T_ + t0;
            *reinterpret_cast<bf16x8*>(dst) = v;
        }
    } else {
        __bf16* __restrict__ out = (mat == 0) ? Qb : Kb;
#pragma unroll
        for (int nt = 0; nt < 8; ++nt)
#pragma unroll
            for (int r = 0; r < 8; ++r)
                out[(size_t)(row0 + half * 8 + r) * DOUT + nt * 16 + lrow] =
                    (__bf16)acc[nt][r];
    }
}

// ---------------------------------------------------------------------------
// Pass 2: causal flash attention.
//   block = 128 threads = 4 waves; each wave owns one 16-row Q tile.
//   grid  = (T/64, B).
// ---------------------------------------------------------------------------
__global__ __launch_bounds__(128)
void attn_kernel(const __bf16* __restrict__ Qb,
                 const __bf16* __restrict__ Kb,
                 const __bf16* __restrict__ Vt,
                 float* __restrict__ out) {
    __shared__ __bf16 plds[4][16][32];   // per-wave P relayout slab (C -> A layout)

    const int lane = threadIdx.x & 31;
    const int wave = threadIdx.x >> 5;
    const int b    = blockIdx.y;
    const int qb   = (blockIdx.x * 4 + wave) * 16;
    const int half = lane >> 4;
    const int lrow = lane & 15;
    const int koff = half ? 8 : 0;
    const float scale = 0.08838834764831845f;   // 1/sqrt(128)

    // Preload Q A-fragments for all 4 d-chunks (d = 128 -> 4 x 32)
    bf16x16 aQ[4];
    const __bf16* qrow = Qb + ((size_t)b * T_ + qb + lrow) * DOUT;
#pragma unroll
    for (int dc = 0; dc < 4; ++dc) {
        const __bf16* p = qrow + dc * 32 + koff;
#pragma unroll
        for (int i = 0; i < 8; ++i) {
            aQ[dc][i]     = p[i];
            aQ[dc][8 + i] = p[16 + i];
        }
    }

    float m_i[8], l_i[8];
    f32x8 o[8];
#pragma unroll
    for (int r = 0; r < 8; ++r) { m_i[r] = -1e30f; l_i[r] = 0.0f; }
#pragma unroll
    for (int nt = 0; nt < 8; ++nt)
#pragma unroll
        for (int r = 0; r < 8; ++r) o[nt][r] = 0.0f;

    // causal: only k-blocks with first column <= last query row
    for (int kb = 0; kb <= qb + 15; kb += 32) {
        // ---- S = (Q K^T) * scale, two 16x16 tiles ----
        f32x8 s[2];
#pragma unroll
        for (int st = 0; st < 2; ++st) {
            f32x8 acc = {0, 0, 0, 0, 0, 0, 0, 0};
            const __bf16* krow = Kb + ((size_t)b * T_ + kb + st * 16 + lrow) * DOUT;
#pragma unroll
            for (int dc = 0; dc < 4; ++dc) {
                bf16x16 bk;
                const __bf16* p = krow + dc * 32 + koff;
#pragma unroll
                for (int i = 0; i < 8; ++i) {
                    bk[i]     = p[i];
                    bk[8 + i] = p[16 + i];
                }
                acc = WMMA_BF16(aQ[dc], bk, acc);
            }
            const int col = kb + st * 16 + lrow;
#pragma unroll
            for (int r = 0; r < 8; ++r) {
                const int row = qb + half * 8 + r;
                const float v = acc[r] * scale;
                s[st][r] = (col <= row) ? v : -1e30f;
            }
        }

        // ---- online softmax (row spans one 16-lane half) ----
        float mtot[8], alpha[8];
#pragma unroll
        for (int r = 0; r < 8; ++r) {
            float v = fmaxf(s[0][r], s[1][r]);
            v = fmaxf(v, __shfl_xor(v, 1, 32));
            v = fmaxf(v, __shfl_xor(v, 2, 32));
            v = fmaxf(v, __shfl_xor(v, 4, 32));
            v = fmaxf(v, __shfl_xor(v, 8, 32));
            mtot[r]  = fmaxf(m_i[r], v);
            alpha[r] = __expf(m_i[r] - mtot[r]);
            m_i[r]   = mtot[r];
        }
#pragma unroll
        for (int r = 0; r < 8; ++r) {
            s[0][r] = __expf(s[0][r] - mtot[r]);
            s[1][r] = __expf(s[1][r] - mtot[r]);
            float v = s[0][r] + s[1][r];
            v += __shfl_xor(v, 1, 32);
            v += __shfl_xor(v, 2, 32);
            v += __shfl_xor(v, 4, 32);
            v += __shfl_xor(v, 8, 32);
            l_i[r] = l_i[r] * alpha[r] + v;
        }
#pragma unroll
        for (int nt = 0; nt < 8; ++nt)
#pragma unroll
            for (int r = 0; r < 8; ++r) o[nt][r] *= alpha[r];

        // ---- relayout P: C-layout -> LDS -> A-fragment (wave-internal) ----
#pragma unroll
        for (int st = 0; st < 2; ++st)
#pragma unroll
            for (int r = 0; r < 8; ++r)
                plds[wave][half * 8 + r][st * 16 + lrow] = (__bf16)s[st][r];

        bf16x16 aP;
        {
            const __bf16* p = &plds[wave][lrow][koff];
#pragma unroll
            for (int i = 0; i < 8; ++i) {
                aP[i]     = p[i];
                aP[8 + i] = p[16 + i];
            }
        }

        // ---- O += P * V  (Vt gives contiguous B-fragments along t) ----
#pragma unroll
        for (int nt = 0; nt < 8; ++nt) {
            bf16x16 bv;
            const __bf16* p = Vt + ((size_t)b * DOUT + nt * 16 + lrow) * T_ + kb + koff;
#pragma unroll
            for (int i = 0; i < 8; ++i) {
                bv[i]     = p[i];
                bv[8 + i] = p[16 + i];
            }
            o[nt] = WMMA_BF16(aP, bv, o[nt]);
        }
    }

    // ---- epilogue: normalize and store f32 ----
    float inv[8];
#pragma unroll
    for (int r = 0; r < 8; ++r) inv[r] = 1.0f / l_i[r];
#pragma unroll
    for (int nt = 0; nt < 8; ++nt)
#pragma unroll
        for (int r = 0; r < 8; ++r)
            out[((size_t)b * T_ + qb + half * 8 + r) * DOUT + nt * 16 + lrow] =
                o[nt][r] * inv[r];
}

// ---------------------------------------------------------------------------
extern "C" void kernel_launch(void* const* d_in, const int* in_sizes, int n_in,
                              void* d_out, int out_size, void* d_ws, size_t ws_size,
                              hipStream_t stream) {
    const float* x  = (const float*)d_in[0];
    const float* Wq = (const float*)d_in[1];
    const float* Wk = (const float*)d_in[2];
    const float* Wv = (const float*)d_in[3];

    // workspace layout (bytes)
    const size_t qkv_bytes = (size_t)BT * DOUT * sizeof(__bf16);   // 4 MiB each
    char* ws = (char*)d_ws;
    __bf16* Qb = (__bf16*)(ws);
    __bf16* Kb = (__bf16*)(ws + qkv_bytes);
    __bf16* Vt = (__bf16*)(ws + 2 * qkv_bytes);
    __bf16* Wt = (__bf16*)(ws + 3 * qkv_bytes);
    const size_t need = 3 * qkv_bytes + (size_t)3 * DIN * DOUT * sizeof(__bf16);
    if (ws_size < need) return;   // deterministic no-op if scratch too small

    // Pass 0: weight transpose+convert
    {
        const int n = 3 * DIN * DOUT;
        prep_weights_kernel<<<(n + 255) / 256, 256, 0, stream>>>(Wq, Wk, Wv, Wt);
    }
    // Pass 1: QKV projections
    qkv_gemm_kernel<<<dim3(BT / 128, 3), 256, 0, stream>>>(x, Wt, Qb, Kb, Vt);
    // Pass 2: causal flash attention
    attn_kernel<<<dim3(T_ / 64, B_), 128, 0, stream>>>(Qb, Kb, Vt, (float*)d_out);
}
